// SelfAttention_84207128805577
// MI455X (gfx1250) — 16-bit and 8-bit matrix operand order repaired and checked by reading, compiled, not run on hardware
//
#include <hip/hip_runtime.h>
#include <hip/hip_bf16.h>
#include <stdint.h>

typedef __attribute__((ext_vector_type(16))) _Float16 v16h;
typedef __attribute__((ext_vector_type(8)))  float    v8f;
typedef __attribute__((ext_vector_type(8)))  int      v8i;

#define EB    2
#define NSEQ  2048
#define EDIM  1024
#define HQ    16
#define HKV   8
#define DHEAD 64
#define KVE   512
#define NTOK  (EB*NSEQ)
#define EPSF  1e-5f

__device__ __forceinline__ unsigned pkh2(float a, float b) {
  union { _Float16 h[2]; unsigned u; } c;
  c.h[0] = (_Float16)a; c.h[1] = (_Float16)b;
  return c.u;
}

__global__ void k_wabs_part(const float* __restrict__ w, int n, float* __restrict__ part) {
  __shared__ float sm[256];
  float s = 0.f;
  for (int i = blockIdx.x * 256 + threadIdx.x; i < n; i += 256 * 256)
    s += fabsf(w[i]);
  sm[threadIdx.x] = s;
  __syncthreads();
  for (int st = 128; st > 0; st >>= 1) {
    if (threadIdx.x < st) sm[threadIdx.x] += sm[threadIdx.x + st];
    __syncthreads();
  }
  if (threadIdx.x == 0) part[blockIdx.x] = sm[0];
}

__global__ void k_wabs_fin(const float* __restrict__ part, float* __restrict__ scales) {
  const int sizes[4] = {EDIM * EDIM, KVE * EDIM, KVE * EDIM, EDIM * EDIM};
  int mat = blockIdx.x;
  __shared__ float sm[256];
  sm[threadIdx.x] = part[mat * 256 + threadIdx.x];
  __syncthreads();
  for (int st = 128; st > 0; st >>= 1) {
    if (threadIdx.x < st) sm[threadIdx.x] += sm[threadIdx.x + st];
    __syncthreads();
  }
  if (threadIdx.x == 0) {
    float mean = sm[0] / (float)sizes[mat];
    float cm = fmaxf(mean, 1e-5f);
    scales[mat]     = cm;
    scales[4 + mat] = 1.0f / cm;
  }
}

__global__ void k_wquant(const float* __restrict__ w, int n,
                         const float* __restrict__ scales, int mat,
                         signed char* __restrict__ wq) {
  int i = blockIdx.x * 256 + threadIdx.x;
  if (i >= n) return;
  float r = rintf(w[i] * scales[4 + mat]);
  r = fminf(fmaxf(r, -1.f), 1.f);
  wq[i] = (signed char)(int)r;
}

__global__ void k_actquant(const float* __restrict__ x, signed char* __restrict__ q,
                           float* __restrict__ inv) {
  int t = blockIdx.x, tid = threadIdx.x;
  const float* xr = x + (size_t)t * EDIM;
  float4 v = ((const float4*)xr)[tid];
  __shared__ float s1[256], s2[256];
  s1[tid] = v.x * v.x + v.y * v.y + v.z * v.z + v.w * v.w;
  s2[tid] = fmaxf(fmaxf(fabsf(v.x), fabsf(v.y)), fmaxf(fabsf(v.z), fabsf(v.w)));
  __syncthreads();
  for (int st = 128; st > 0; st >>= 1) {
    if (tid < st) { s1[tid] += s1[tid + st]; s2[tid] = fmaxf(s2[tid], s2[tid + st]); }
    __syncthreads();
  }
  float r = rsqrtf(s1[0] / (float)EDIM + EPSF);
  float amax = s2[0] * r;
  float scale = 127.f / fmaxf(amax, 1e-5f);
  float m = r * scale;
  char4 o;
  o.x = (signed char)(int)fminf(fmaxf(rintf(v.x * m), -128.f), 127.f);
  o.y = (signed char)(int)fminf(fmaxf(rintf(v.y * m), -128.f), 127.f);
  o.z = (signed char)(int)fminf(fmaxf(rintf(v.z * m), -128.f), 127.f);
  o.w = (signed char)(int)fminf(fmaxf(rintf(v.w * m), -128.f), 127.f);
  ((char4*)(q + (size_t)t * EDIM))[tid] = o;
  if (tid == 0) inv[t] = 1.0f / scale;
}

__global__ void k_gemm_iu8(const signed char* __restrict__ A, const signed char* __restrict__ W,
                           const float* __restrict__ ainv, const float* __restrict__ scales, int mat,
                           const float* __restrict__ bias, int M, int Nout, int K,
                           int mode, float qmul, char* __restrict__ out) {
  int wv = blockIdx.x * 4 + (threadIdx.x >> 5);
  int lane = threadIdx.x & 31;
  int tilesN = Nout >> 6;
  int tiles = (M >> 4) * tilesN;
  if (wv >= tiles) return;
  int tm = (wv / tilesN) << 4;
  int tn = (wv % tilesN) << 6;
  int lhi = lane >> 4, llo = lane & 15;

  const signed char* ar = A + (size_t)(tm + llo) * K + lhi * 16;
  const signed char* br0 = W + (size_t)(tn + llo) * K + lhi * 16;
  const signed char* br1 = br0 + (size_t)16 * K;
  const signed char* br2 = br0 + (size_t)32 * K;
  const signed char* br3 = br0 + (size_t)48 * K;

  v8i acc[4] = {};
  for (int k0 = 0; k0 < K; k0 += 64) {
    union { v8i v; int2 d[4]; } a;
    a.d[0] = *(const int2*)(ar + k0);
    a.d[1] = *(const int2*)(ar + k0 + 8);
    a.d[2] = *(const int2*)(ar + k0 + 32);
    a.d[3] = *(const int2*)(ar + k0 + 40);
    union { v8i v; int4 d[2]; } b;
    b.d[0] = *(const int4*)(br0 + k0);
    b.d[1] = *(const int4*)(br0 + k0 + 32);
    acc[0] = __builtin_amdgcn_wmma_i32_16x16x64_iu8(true, a.v, true, b.v, acc[0], false, false);
    b.d[0] = *(const int4*)(br1 + k0);
    b.d[1] = *(const int4*)(br1 + k0 + 32);
    acc[1] = __builtin_amdgcn_wmma_i32_16x16x64_iu8(true, a.v, true, b.v, acc[1], false, false);
    b.d[0] = *(const int4*)(br2 + k0);
    b.d[1] = *(const int4*)(br2 + k0 + 32);
    acc[2] = __builtin_amdgcn_wmma_i32_16x16x64_iu8(true, a.v, true, b.v, acc[2], false, false);
    b.d[0] = *(const int4*)(br3 + k0);
    b.d[1] = *(const int4*)(br3 + k0 + 32);
    acc[3] = __builtin_amdgcn_wmma_i32_16x16x64_iu8(true, a.v, true, b.v, acc[3], false, false);
  }

  float wmean = scales[mat];
  int rowb = tm + lhi * 8;
  float4 av0 = *(const float4*)(ainv + rowb);
  float4 av1 = *(const float4*)(ainv + rowb + 4);
  float avv[8] = {av0.x, av0.y, av0.z, av0.w, av1.x, av1.y, av1.z, av1.w};

#pragma unroll
  for (int j = 0; j < 4; ++j) {
    int col = tn + j * 16 + llo;
    float bcol = bias[col];
    float f[8];
#pragma unroll
    for (int v = 0; v < 8; ++v)
      f[v] = ((float)acc[j][v] * avv[v] * wmean + bcol) * qmul;

    if (mode == 0 || mode == 1) {
      _Float16* o = (_Float16*)out;
#pragma unroll
      for (int v = 0; v < 8; ++v)
        o[(size_t)(rowb + v) * Nout + col] = (_Float16)f[v];
    } else if (mode == 2) {
      union { int4 i; _Float16 h[8]; } pk;
#pragma unroll
      for (int v = 0; v < 8; ++v) pk.h[v] = (_Float16)f[v];
      int bb = rowb / NSEQ, n = rowb % NSEQ;
      int hk = col >> 6, dd = col & 63;
      _Float16* o = (_Float16*)out + ((size_t)(bb * HKV + hk) * DHEAD + dd) * NSEQ + n;
      *(int4*)o = pk.i;
    } else {
      float* o = (float*)out;
#pragma unroll
      for (int v = 0; v < 8; ++v)
        o[(size_t)(rowb + v) * Nout + col] = f[v];
    }
  }
}

__global__ void k_attn(const _Float16* __restrict__ Q, const _Float16* __restrict__ Kk,
                       const _Float16* __restrict__ Vt, float* __restrict__ O) {
  int wv = blockIdx.x * 4 + (threadIdx.x >> 5);
  int lane = threadIdx.x & 31;
  int qt2 = wv & 63;
  int h   = (wv >> 6) & 15;
  int b   = wv >> 10;
  int hk  = h >> 1;
  int n0  = qt2 << 5;
  int lhi = lane >> 4, llo = lane & 15;

  union QF { v16h v; int4 d[2]; };
  QF qf[2][2];
#pragma unroll
  for (int qt = 0; qt < 2; ++qt) {
    const _Float16* qbase = Q + ((size_t)(b * NSEQ + n0 + qt * 16 + llo)) * EDIM + h * DHEAD + lhi * 8;
    qf[qt][0].d[0] = *(const int4*)(qbase);
    qf[qt][0].d[1] = *(const int4*)(qbase + 16);
    qf[qt][1].d[0] = *(const int4*)(qbase + 32);
    qf[qt][1].d[1] = *(const int4*)(qbase + 48);
  }

  const _Float16* kbase = Kk + (size_t)b * NSEQ * KVE + hk * DHEAD;
  const _Float16* vbase = Vt + ((size_t)(b * HKV + hk)) * DHEAD * NSEQ;

  v8f oa[2][4] = {};
  float mrun[2] = {-1e30f, -1e30f};
  float lrun[2] = {0.f, 0.f};

  for (int s0 = 0; s0 < NSEQ; s0 += 32) {
    QF ka[4];
    {
      const _Float16* kr0 = kbase + (size_t)(s0 + llo) * KVE + lhi * 8;
      ka[0].d[0] = *(const int4*)(kr0);
      ka[0].d[1] = *(const int4*)(kr0 + 16);
      ka[1].d[0] = *(const int4*)(kr0 + 32);
      ka[1].d[1] = *(const int4*)(kr0 + 48);
      const _Float16* kr1 = kbase + (size_t)(s0 + 16 + llo) * KVE + lhi * 8;
      ka[2].d[0] = *(const int4*)(kr1);
      ka[2].d[1] = *(const int4*)(kr1 + 16);
      ka[3].d[0] = *(const int4*)(kr1 + 32);
      ka[3].d[1] = *(const int4*)(kr1 + 48);
    }

    v16h pb[2];
#pragma unroll
    for (int qt = 0; qt < 2; ++qt) {
      v8f sc0 = {}, sc1 = {};
      sc0 = __builtin_amdgcn_wmma_f32_16x16x32_f16(false, ka[0].v, false, qf[qt][0].v, (short)0, sc0, false, false);
      sc0 = __builtin_amdgcn_wmma_f32_16x16x32_f16(false, ka[1].v, false, qf[qt][1].v, (short)0, sc0, false, false);
      sc1 = __builtin_amdgcn_wmma_f32_16x16x32_f16(false, ka[2].v, false, qf[qt][0].v, (short)0, sc1, false, false);
      sc1 = __builtin_amdgcn_wmma_f32_16x16x32_f16(false, ka[3].v, false, qf[qt][1].v, (short)0, sc1, false, false);

      float cmax = -1e30f;
#pragma unroll
      for (int v = 0; v < 8; ++v) cmax = fmaxf(cmax, fmaxf(sc0[v], sc1[v]));
      cmax = fmaxf(cmax, __shfl_xor(cmax, 16, 32));
      float mnew = fmaxf(mrun[qt], cmax);
      float corr = __expf(mrun[qt] - mnew);
      float p0[8], p1[8];
      float lsum = 0.f;
#pragma unroll
      for (int v = 0; v < 8; ++v) {
        p0[v] = __expf(sc0[v] - mnew);
        p1[v] = __expf(sc1[v] - mnew);
        lsum += p0[v] + p1[v];
      }
      lsum += __shfl_xor(lsum, 16, 32);
      lrun[qt] = lrun[qt] * corr + lsum;
      mrun[qt] = mnew;
#pragma unroll
      for (int v = 0; v < 8; ++v) {
        oa[qt][0][v] *= corr; oa[qt][1][v] *= corr;
        oa[qt][2][v] *= corr; oa[qt][3][v] *= corr;
      }

      unsigned a0[4], a1[4];
#pragma unroll
      for (int i = 0; i < 4; ++i) {
        a0[i] = pkh2(p0[2 * i], p0[2 * i + 1]);
        a1[i] = pkh2(p1[2 * i], p1[2 * i + 1]);
      }
      union { v16h v; unsigned u[8]; } pbu;
      bool low = (lhi == 0);
#pragma unroll
      for (int i = 0; i < 4; ++i) {
        unsigned b0 = (unsigned)__shfl_xor((int)a0[i], 16, 32);
        unsigned b1 = (unsigned)__shfl_xor((int)a1[i], 16, 32);
        pbu.u[i]     = low ? a0[i] : b1;
        pbu.u[4 + i] = low ? b0 : a1[i];
      }
      pb[qt] = pbu.v;
    }

#pragma unroll
    for (int t = 0; t < 4; ++t) {
      const _Float16* vr = vbase + (size_t)(16 * t + llo) * NSEQ + s0 + lhi * 16;
      QF va;
      va.d[0] = *(const int4*)(vr);
      va.d[1] = *(const int4*)(vr + 8);
      oa[0][t] = __builtin_amdgcn_wmma_f32_16x16x32_f16(false, va.v, false, pb[0], (short)0, oa[0][t], false, false);
      oa[1][t] = __builtin_amdgcn_wmma_f32_16x16x32_f16(false, va.v, false, pb[1], (short)0, oa[1][t], false, false);
    }
  }

#pragma unroll
  for (int qt = 0; qt < 2; ++qt) {
    float linv = 1.0f / lrun[qt];
    float* ob = O + ((size_t)(b * NSEQ + n0 + qt * 16 + llo)) * EDIM + h * DHEAD + lhi * 8;
#pragma unroll
    for (int t = 0; t < 4; ++t) {
      float4 f0 = make_float4(oa[qt][t][0] * linv, oa[qt][t][1] * linv, oa[qt][t][2] * linv, oa[qt][t][3] * linv);
      float4 f1 = make_float4(oa[qt][t][4] * linv, oa[qt][t][5] * linv, oa[qt][t][6] * linv, oa[qt][t][7] * linv);
      *(float4*)(ob + 16 * t) = f0;
      *(float4*)(ob + 16 * t + 4) = f1;
    }
  }
}

__global__ void k_ln_quant(const float* __restrict__ x, const float* __restrict__ g,
                           const float* __restrict__ bb, signed char* __restrict__ q,
                           float* __restrict__ inv) {
  int t = blockIdx.x, tid = threadIdx.x;
  const float* xr = x + (size_t)t * EDIM;
  float4 v = ((const float4*)xr)[tid];
  __shared__ float s1[256], s2[256];
  s1[tid] = v.x + v.y + v.z + v.w;
  s2[tid] = v.x * v.x + v.y * v.y + v.z * v.z + v.w * v.w;
  __syncthreads();
  for (int st = 128; st > 0; st >>= 1) {
    if (tid < st) { s1[tid] += s1[tid + st]; s2[tid] += s2[tid + st]; }
    __syncthreads();
  }
  float mean = s1[0] / (float)EDIM;
  float var  = s2[0] / (float)EDIM - mean * mean;
  float rstd = rsqrtf(var + EPSF);
  float4 gg = ((const float4*)g)[tid];
  float4 bv = ((const float4*)bb)[tid];
  float y0 = (v.x - mean) * rstd * gg.x + bv.x;
  float y1 = (v.y - mean) * rstd * gg.y + bv.y;
  float y2 = (v.z - mean) * rstd * gg.z + bv.z;
  float y3 = (v.w - mean) * rstd * gg.w + bv.w;
  __syncthreads();
  s1[tid] = y0 * y0 + y1 * y1 + y2 * y2 + y3 * y3;
  s2[tid] = fmaxf(fmaxf(fabsf(y0), fabsf(y1)), fmaxf(fabsf(y2), fabsf(y3)));
  __syncthreads();
  for (int st = 128; st > 0; st >>= 1) {
    if (tid < st) { s1[tid] += s1[tid + st]; s2[tid] = fmaxf(s2[tid], s2[tid + st]); }
    __syncthreads();
  }
  float r = rsqrtf(s1[0] / (float)EDIM + EPSF);
  float amax = s2[0] * r;
  float scale = 127.f / fmaxf(amax, 1e-5f);
  float m = r * scale;
  char4 o;
  o.x = (signed char)(int)fminf(fmaxf(rintf(y0 * m), -128.f), 127.f);
  o.y = (signed char)(int)fminf(fmaxf(rintf(y1 * m), -128.f), 127.f);
  o.z = (signed char)(int)fminf(fmaxf(rintf(y2 * m), -128.f), 127.f);
  o.w = (signed char)(int)fminf(fmaxf(rintf(y3 * m), -128.f), 127.f);
  ((char4*)(q + (size_t)t * EDIM))[tid] = o;
  if (tid == 0) inv[t] = 1.0f / scale;
}

extern "C" void kernel_launch(void* const* d_in, const int* in_sizes, int n_in,
                              void* d_out, int out_size, void* d_ws, size_t ws_size,
                              hipStream_t stream) {
  const float* x   = (const float*)d_in[0];
  const float* wq  = (const float*)d_in[1];
  const float* bq  = (const float*)d_in[2];
  const float* wk  = (const float*)d_in[3];
  const float* bk  = (const float*)d_in[4];
  const float* wv  = (const float*)d_in[5];
  const float* bv  = (const float*)d_in[6];
  const float* wo  = (const float*)d_in[7];
  const float* bo  = (const float*)d_in[8];
  const float* lng = (const float*)d_in[9];
  const float* lnb = (const float*)d_in[10];

  char* ws = (char*)d_ws;
  size_t off = 0;
  auto alloc = [&](size_t bytes) { size_t o = off; off = (off + bytes + 255) & ~(size_t)255; return o; };
  float*       scales = (float*)(ws + alloc(8 * sizeof(float)));
  float*       part   = (float*)(ws + alloc(4 * 256 * sizeof(float)));
  signed char* wq8    = (signed char*)(ws + alloc((size_t)EDIM * EDIM));
  signed char* wk8    = (signed char*)(ws + alloc((size_t)KVE * EDIM));
  signed char* wv8    = (signed char*)(ws + alloc((size_t)KVE * EDIM));
  signed char* wo8    = (signed char*)(ws + alloc((size_t)EDIM * EDIM));
  signed char* xq1    = (signed char*)(ws + alloc((size_t)NTOK * EDIM));
  float*       inv1   = (float*)(ws + alloc((size_t)NTOK * 4));
  _Float16*    qf     = (_Float16*)(ws + alloc((size_t)NTOK * EDIM * 2));
  _Float16*    kf     = (_Float16*)(ws + alloc((size_t)NTOK * KVE * 2));
  _Float16*    vt     = (_Float16*)(ws + alloc((size_t)NTOK * KVE * 2));
  float*       attn   = (float*)(ws + alloc((size_t)NTOK * EDIM * 4));
  signed char* xq2    = (signed char*)(ws + alloc((size_t)NTOK * EDIM));
  float*       inv2   = (float*)(ws + alloc((size_t)NTOK * 4));
  (void)ws_size; (void)in_sizes; (void)n_in; (void)out_size;

  k_wabs_part<<<256, 256, 0, stream>>>(wq, EDIM * EDIM, part + 0 * 256);
  k_wabs_part<<<256, 256, 0, stream>>>(wk, KVE * EDIM, part + 1 * 256);
  k_wabs_part<<<256, 256, 0, stream>>>(wv, KVE * EDIM, part + 2 * 256);
  k_wabs_part<<<256, 256, 0, stream>>>(wo, EDIM * EDIM, part + 3 * 256);
  k_wabs_fin<<<4, 256, 0, stream>>>(part, scales);
  k_wquant<<<(EDIM * EDIM) / 256, 256, 0, stream>>>(wq, EDIM * EDIM, scales, 0, wq8);
  k_wquant<<<(KVE * EDIM) / 256, 256, 0, stream>>>(wk, KVE * EDIM, scales, 1, wk8);
  k_wquant<<<(KVE * EDIM) / 256, 256, 0, stream>>>(wv, KVE * EDIM, scales, 2, wv8);
  k_wquant<<<(EDIM * EDIM) / 256, 256, 0, stream>>>(wo, EDIM * EDIM, scales, 3, wo8);

  k_actquant<<<NTOK, 256, 0, stream>>>(x, xq1, inv1);

  int blkq = (NTOK / 16) * (EDIM / 64) / 4;
  int blkk = (NTOK / 16) * (KVE / 64) / 4;
  k_gemm_iu8<<<blkq, 128, 0, stream>>>(xq1, wq8, inv1, scales, 0, bq, NTOK, EDIM, EDIM, 0, 0.125f, (char*)qf);
  k_gemm_iu8<<<blkk, 128, 0, stream>>>(xq1, wk8, inv1, scales, 1, bk, NTOK, KVE, EDIM, 1, 1.0f, (char*)kf);
  k_gemm_iu8<<<blkk, 128, 0, stream>>>(xq1, wv8, inv1, scales, 2, bv, NTOK, KVE, EDIM, 2, 1.0f, (char*)vt);

  k_attn<<<(EB * HQ * (NSEQ / 32)) / 4, 128, 0, stream>>>(qf, kf, vt, attn);

  k_ln_quant<<<NTOK, 256, 0, stream>>>(attn, lng, lnb, xq2, inv2);
  k_gemm_iu8<<<blkq, 128, 0, stream>>>(xq2, wo8, inv2, scales, 3, bo, NTOK, EDIM, EDIM, 3, 1.0f, (char*)d_out);
}
